// ConditionalSoftmax_18330920419356
// MI455X (gfx1250) — matrix operand order repaired and checked by reading, not compiled, not run on hardware
//
#include <hip/hip_runtime.h>

// Problem constants (match reference)
#define B_ROWS 8192
#define D_IN   300
#define D_PAD  320            // K padded to multiple of 32 for GEMM1
#define H_DIM  512
#define V_SUP  20000
#define NTILE  (V_SUP / 16)   // 1250 column tiles in GEMM2
#define KS1    (D_PAD / 32)   // 10 k-steps in GEMM1
#define KS2    (H_DIM / 32)   // 16 k-steps in GEMM2
#define RT1    (B_ROWS / 16)  // 512 row tiles

// Fragment stream: one 16x32 bf16 fragment = 32 lanes x 16 u16 = 1024 bytes.
// Per-lane data is 32 contiguous bytes -> a single v8u (2 x b128) load.

typedef __attribute__((ext_vector_type(16))) __bf16       v16bf;
typedef __attribute__((ext_vector_type(8)))  float        v8f;
typedef __attribute__((ext_vector_type(8)))  unsigned int v8u;

__device__ __forceinline__ unsigned short f2bf(float x) {
  unsigned int u = __builtin_bit_cast(unsigned int, x);
  return (unsigned short)((u + 0x7FFFu + ((u >> 16) & 1u)) >> 16); // RNE
}

// ===================== prep kernels (emit fragment-major layouts) =====================

// A-fragment slot (lane l, slot s) -> (m_in, k_local), ISA 7.12.2 16-bit A 16x32:
//   m_in = l&15, half = l>>4, v = s>>1, p = s&1
//   kb = (v<4) ? 8*half+2v : 16+8*half+2(v-4);  k_local = kb + p
// B-fragment slot (lane l, slot s) -> (k_local, n_in):
//   n_in = l&15, kh = l>>4, v = s>>1, p = s&1;  k_local = ((v<4) ? 8*kh+2v : 16+8*kh+2(v-4)) + p

__global__ void prep_afrag(const int* __restrict__ cs, const float* __restrict__ vectors,
                           unsigned short* __restrict__ Af) {
  int i = blockIdx.x * 256 + threadIdx.x;            // over RT1*KS1*512 u16 slots
  if (i >= RT1 * KS1 * 512) return;
  int s = i & 15, l = (i >> 4) & 31, f = i >> 9;
  int ks = f % KS1, rt = f / KS1;
  int m = rt * 16 + (l & 15);
  int half = l >> 4, v = s >> 1, p = s & 1;
  int kb = (v < 4) ? (8 * half + 2 * v) : (16 + 8 * half + 2 * (v - 4));
  int k = ks * 32 + kb + p;
  float val = (k < D_IN) ? vectors[(size_t)cs[m] * D_IN + k] : 0.0f;
  Af[i] = f2bf(val);
}

__global__ void prep_tgt(const int* __restrict__ ws_idx, const int* __restrict__ v2s,
                         int* __restrict__ tgt) {
  int b = blockIdx.x * 256 + threadIdx.x;
  if (b < B_ROWS) tgt[b] = v2s[ws_idx[b]];
}

// generic B-fragment converter: W row-major [K, ld], k padded to KS*32 with zeros
__global__ void prep_bfrag(const float* __restrict__ W, unsigned short* __restrict__ Bf,
                           int KS, int ld, int kmax, int total) {
  int i = blockIdx.x * 256 + threadIdx.x;            // over (NCT*KS)*512 u16 slots
  if (i >= total) return;
  int s = i & 15, l = (i >> 4) & 31, f = i >> 9;
  int ks = f % KS, ct = f / KS;
  int n = ct * 16 + (l & 15);
  int kh = l >> 4, v = s >> 1, p = s & 1;
  int k = ks * 32 + ((v < 4) ? (8 * kh + 2 * v) : (16 + 8 * kh + 2 * (v - 4))) + p;
  float val = (k < kmax) ? W[(size_t)k * ld + n] : 0.0f;
  Bf[i] = f2bf(val);
}

// ===================== GEMM1: h = relu(A @ W1 + b1) -> Hf (GEMM2 A-fragment layout) ====
// 8 waves/WG, one 16x16 tile per wave; 16384 tiles -> 2048 WGs.
__global__ void gemm1_relu(const unsigned short* __restrict__ Af,
                           const unsigned short* __restrict__ Bf1,
                           const float* __restrict__ b1,
                           unsigned short* __restrict__ Hf) {
  int wave = threadIdx.x >> 5;
  int tile = blockIdx.x * 8 + wave;
  int rt = tile >> 5;   // row tile 0..511
  int ct = tile & 31;   // col tile 0..31
  int lane = threadIdx.x & 31;
  int n = lane & 15, half = lane >> 4;

  v8f acc = {};
#pragma unroll
  for (int ks = 0; ks < KS1; ++ks) {
    v8u ua = *(const v8u*)(Af + ((size_t)(rt * KS1 + ks) << 9) + (lane << 4));
    v8u ub = *(const v8u*)(Bf1 + ((size_t)(ct * KS1 + ks) << 9) + (lane << 4));
    v16bf a = __builtin_bit_cast(v16bf, ua);
    v16bf b = __builtin_bit_cast(v16bf, ub);
    acc = __builtin_amdgcn_wmma_f32_16x16x32_bf16(false, a, false, b, (short)0, acc,
                                                  false, false);
  }
  float bias = b1[ct * 16 + n];

  // scatter into Hf = GEMM2 A-fragment layout: h element (row, hcol):
  //   ks2 = hcol>>5, c = hcol&31, half2 = (c>>3)&1, p = c&1,
  //   v2 = (c<16) ? (c&7)>>1 : 4+((c&7)>>1), lane2 = half2*16 + (row&15)
  int hcol = ct * 16 + n;
  int ks2 = hcol >> 5, c = hcol & 31;
  int half2 = (c >> 3) & 1, p = c & 1;
  int v2 = (c < 16) ? ((c & 7) >> 1) : (4 + ((c & 7) >> 1));
  int base = ((rt * KS2 + ks2) << 9) + ((half2 << 4) << 4) + (v2 << 1) + p;
#pragma unroll
  for (int v = 0; v < 8; ++v) {
    float x = fmaxf(acc[v] + bias, 0.0f);
    int m_in = v + 8 * half;                 // row & 15
    Hf[base + (m_in << 4)] = f2bf(x);
  }
}

// ===================== GEMM2 + fused log-softmax NLL =====================
// WG owns 64 rows; stages its 64 A-fragments (64 KB, contiguous) in LDS.
// 8 waves stride over 1250 column tiles; online (max,sumexp) + target capture.
__global__ void gemm2_nll(const unsigned short* __restrict__ Hf,
                          const unsigned short* __restrict__ Bf2,
                          const float* __restrict__ b2,
                          const int* __restrict__ tgt,
                          float* __restrict__ out) {
  __shared__ __align__(32) unsigned short As[64 * 512];   // 64 fragments, 64 KB

  int r0 = blockIdx.x * 64;

  // contiguous 128-bit staged copy of 64 fragments
  {
    const uint4* src = (const uint4*)Hf + (size_t)blockIdx.x * 4096;
    uint4* dst = (uint4*)As;
    for (int t = threadIdx.x; t < 4096; t += 256) dst[t] = src[t];
  }
  __syncthreads();

  int wave = threadIdx.x >> 5;
  int lane = threadIdx.x & 31;
  int n = lane & 15, half = lane >> 4;

  // target support index per (row-tile, accumulator row)
  int tg_idx[4][8];
#pragma unroll
  for (int rt = 0; rt < 4; ++rt)
#pragma unroll
    for (int v = 0; v < 8; ++v)
      tg_idx[rt][v] = tgt[r0 + rt * 16 + v + 8 * half];

  float rmax[4][8], rsum[4][8], tlog[4][8];
#pragma unroll
  for (int rt = 0; rt < 4; ++rt)
#pragma unroll
    for (int v = 0; v < 8; ++v) { rmax[rt][v] = -1e30f; rsum[rt][v] = 0.0f; tlog[rt][v] = 0.0f; }

  for (int ct = wave; ct < NTILE; ct += 8) {
    int n0 = ct * 16;
    v8f acc[4] = {};

#pragma unroll
    for (int ks = 0; ks < KS2; ++ks) {
      v8u ub = *(const v8u*)(Bf2 + ((size_t)(ct * KS2 + ks) << 9) + (lane << 4));
      v16bf bf = __builtin_bit_cast(v16bf, ub);
#pragma unroll
      for (int rt = 0; rt < 4; ++rt) {
        v8u ua = *(const v8u*)(As + (((rt * KS2 + ks) << 9) + (lane << 4)));
        v16bf af = __builtin_bit_cast(v16bf, ua);
        acc[rt] = __builtin_amdgcn_wmma_f32_16x16x32_bf16(false, af, false, bf, (short)0,
                                                          acc[rt], false, false);
      }
    }

    // epilogue: bias, target capture, online max/sumexp merge
    float bias = b2[n0 + n];
#pragma unroll
    for (int rt = 0; rt < 4; ++rt)
#pragma unroll
      for (int v = 0; v < 8; ++v) {
        float x = acc[rt][v] + bias;
        if (n0 + n == tg_idx[rt][v]) tlog[rt][v] += x;
        float nm = fmaxf(rmax[rt][v], x);
        rsum[rt][v] = rsum[rt][v] * __expf(rmax[rt][v] - nm) + __expf(x - nm);
        rmax[rt][v] = nm;
      }
  }

  __syncthreads();                       // everyone done reading As
  float* redM = (float*)As;              // reuse LDS: 3 arrays of [8][64] floats
  float* redS = redM + 512;
  float* redT = redM + 1024;

  // intra-wave reduce across the 16-lane group holding one row's 16 columns
#pragma unroll
  for (int rt = 0; rt < 4; ++rt)
#pragma unroll
    for (int v = 0; v < 8; ++v) {
      float m = rmax[rt][v], s = rsum[rt][v], t = tlog[rt][v];
#pragma unroll
      for (int off = 1; off < 16; off <<= 1) {
        float om = __shfl_xor(m, off, 32);
        float os = __shfl_xor(s, off, 32);
        float ot = __shfl_xor(t, off, 32);
        float nm = fmaxf(m, om);
        s = s * __expf(m - nm) + os * __expf(om - nm);
        m = nm;
        t += ot;
      }
      if ((lane & 15) == 0) {
        int lrow = rt * 16 + v + 8 * half;     // 0..63, unique per (rt,v,half)
        redM[wave * 64 + lrow] = m;
        redS[wave * 64 + lrow] = s;
        redT[wave * 64 + lrow] = t;
      }
    }
  __syncthreads();

  // cross-wave reduce: 64 threads, one row each
  if (threadIdx.x < 64) {
    int row = threadIdx.x;
    float m = -1e30f, s = 0.0f, t = 0.0f;
#pragma unroll
    for (int w = 0; w < 8; ++w) {
      float om = redM[w * 64 + row];
      float os = redS[w * 64 + row];
      float nm = fmaxf(m, om);
      s = s * __expf(m - nm) + os * __expf(om - nm);
      m = nm;
      t += redT[w * 64 + row];
    }
    out[r0 + row] = (m + __logf(s)) - t;   // NLL = logsumexp - target_logit
  }
}

// ===================== host-side launch =====================

extern "C" void kernel_launch(void* const* d_in, const int* in_sizes, int n_in,
                              void* d_out, int out_size, void* d_ws, size_t ws_size,
                              hipStream_t stream) {
  const int*   ws_i  = (const int*)d_in[0];   // ws [B] int32
  const int*   cs    = (const int*)d_in[1];   // cs [B] int32
  const float* vec   = (const float*)d_in[2]; // vectors [50000,300]
  const float* W1    = (const float*)d_in[3]; // [300,512]
  const float* b1    = (const float*)d_in[4]; // [512]
  const float* W2    = (const float*)d_in[5]; // [512,20000]
  const float* b2    = (const float*)d_in[6]; // [20000]
  const int*   v2s   = (const int*)d_in[7];   // [50000] int32
  float*       out   = (float*)d_out;         // [B]

  // workspace layout (all offsets 256B-aligned)
  char* wsb = (char*)d_ws;
  size_t off = 0;
  unsigned short* Af  = (unsigned short*)(wsb + off); off += (size_t)RT1 * KS1 * 512 * 2;   //  5,242,880
  unsigned short* Bf1 = (unsigned short*)(wsb + off); off += (size_t)32 * KS1 * 512 * 2;    //    327,680
  unsigned short* Bf2 = (unsigned short*)(wsb + off); off += (size_t)NTILE * KS2 * 512 * 2; // 20,480,000
  unsigned short* Hf  = (unsigned short*)(wsb + off); off += (size_t)RT1 * KS2 * 512 * 2;   //  8,388,608
  int*            tgt = (int*)(wsb + off);            off += (size_t)B_ROWS * 4;            //     32,768
  (void)ws_size; (void)in_sizes; (void)n_in; (void)out_size;

  // 1) gather + bf16 convert embeddings into A-fragment stream; resolve targets
  prep_afrag<<<(RT1 * KS1 * 512 + 255) / 256, 256, 0, stream>>>(cs, vec, Af);
  prep_tgt<<<(B_ROWS + 255) / 256, 256, 0, stream>>>(ws_i, v2s, tgt);

  // 2) weights -> bf16 B-fragment streams (W1 K-padded 300->320)
  prep_bfrag<<<(32 * KS1 * 512 + 255) / 256, 256, 0, stream>>>(W1, Bf1, KS1, H_DIM, D_IN,
                                                               32 * KS1 * 512);
  prep_bfrag<<<(NTILE * KS2 * 512 + 255) / 256, 256, 0, stream>>>(W2, Bf2, KS2, V_SUP, H_DIM,
                                                                  NTILE * KS2 * 512);

  // 3) h = relu(A @ W1 + b1) -> Hf in GEMM2 A-fragment layout
  gemm1_relu<<<(RT1 * 32) / 8, 256, 0, stream>>>(Af, Bf1, b1, Hf);

  // 4) fused h @ W2 + b2 -> online log-softmax -> per-row NLL
  gemm2_nll<<<B_ROWS / 64, 256, 0, stream>>>(Hf, Bf2, b2, tgt, out);
}
